// MultiheadAttention_45913200394349
// MI455X (gfx1250) — matrix operand order repaired and checked by reading, not compiled, not run on hardware
//
#include <hip/hip_runtime.h>

// ---------------------------------------------------------------------------
// MHA forward for MI455X (gfx1250, wave32, WMMA 16x16x32 f16 -> f32)
//   S=2048 tokens, E=1024 embed, H=16 heads, D=64 head dim.
// ---------------------------------------------------------------------------

typedef __attribute__((ext_vector_type(16))) _Float16 v16h;
typedef __attribute__((ext_vector_type(8)))  _Float16 h8;
typedef __attribute__((ext_vector_type(4)))  _Float16 h4;
typedef __attribute__((ext_vector_type(8)))  float    v8f;
typedef __attribute__((ext_vector_type(4)))  float    f4;

union AF { v16h v; h8 h[2]; };

#define SEQ  2048
#define EMB  1024
#define NH   16
#define HD   64

// --- WMMA wrapper: D = A(16x32 f16) x B(32x16 f16) + C(16x16 f32) ----------
__device__ __forceinline__ v8f wmma32(v16h a, v16h b, v8f c) {
  return __builtin_amdgcn_wmma_f32_16x16x32_f16(
      /*neg_a=*/false, a, /*neg_b=*/false, b,
      /*c_mod=*/(short)0, c, /*reuse_a=*/false, /*reuse_b=*/false);
}

// --- 16-lane (half-wave) reductions via DPP16 ROW_XMASK (pure VALU, no LDS).
//     dpp_ctrl 0x160|mask xors the lane id within each 16-lane row. ---------
template <int MASK>
__device__ __forceinline__ float rowxor_f32(float x) {
  return __int_as_float(__builtin_amdgcn_update_dpp(
      0, __float_as_int(x), 0x160 | MASK, 0xF, 0xF, true));
}
__device__ __forceinline__ float rmax16(float x) {
  x = fmaxf(x, rowxor_f32<1>(x));
  x = fmaxf(x, rowxor_f32<2>(x));
  x = fmaxf(x, rowxor_f32<4>(x));
  x = fmaxf(x, rowxor_f32<8>(x));
  return x;
}
__device__ __forceinline__ float rsum16(float x) {
  x += rowxor_f32<1>(x);
  x += rowxor_f32<2>(x);
  x += rowxor_f32<4>(x);
  x += rowxor_f32<8>(x);
  return x;
}

// --- A fragment (16x32, MxK): lane = row M (lane&15); half-wave selects
//     K chunks {half*8..+7} and {16+half*8..+7}   (ISA 7.12.2 16-bit A) -----
__device__ __forceinline__ v16h load_frag_a(const _Float16* base, int strideH, int lane) {
  const int m = lane & 15, hf = lane >> 4;
  const _Float16* rp = base + m * strideH + hf * 8;
  AF f;
  f.h[0] = *(const h8*)(rp);        // K = hf*8 .. +7
  f.h[1] = *(const h8*)(rp + 16);   // K = 16 + hf*8 .. +7
  return f.v;
}

// --- B fragment (32x16, KxN) loaded from a row-major [N][K] array
//     (i.e. B-transposed): lane = col N; half-wave selects K chunks
//     {half*8..+7} and {16+half*8..+7} -----------------------------------
__device__ __forceinline__ v16h load_frag_b(const _Float16* baseT, int strideH, int lane) {
  const int n = lane & 15, hf = lane >> 4;
  const _Float16* rp = baseT + n * strideH + hf * 8;
  AF f;
  f.h[0] = *(const h8*)(rp);        // K = hf*8 .. +7
  f.h[1] = *(const h8*)(rp + 16);   // K = 16 + hf*8 .. +7
  return f.v;
}

// ---------------------------------------------------------------------------
// GEMM: C[2048,1024] = A[2048,1024] * W[1024,1024]^T
//   A_F16=0: A is f32 (converted to f16 through LDS); =1: A already f16.
//   OMODE 0: f16 out, layout [H][S][64], scaled   (Q/K projections)
//   OMODE 1: f16 out, layout [H][64][S] transposed (V projection)
//   OMODE 2: f32 out, layout [S][1024]             (output projection)
// Block: 256 thr (8 waves) -> 128x128 tile; wave = 32x64 (2x4 WMMA tiles).
// ---------------------------------------------------------------------------
#define LDS_STRIDE 40   // halves: 80 bytes/row, 16B aligned, conflict-skewed

template <bool A_F16, int OMODE>
__global__ __launch_bounds__(256)
void gemm_xwT_kernel(const void* __restrict__ Ag, const float* __restrict__ Wg,
                     void* __restrict__ Og, float scale) {
  __shared__ __align__(16) _Float16 sA[128 * LDS_STRIDE];
  __shared__ __align__(16) _Float16 sB[128 * LDS_STRIDE];

  const int tid  = threadIdx.x;
  const int lane = tid & 31, wid = tid >> 5;
  const int wm = wid >> 1;          // 0..3 : wave row  (32 rows each)
  const int wn = wid & 1;           // 0..1 : wave col  (64 cols each)
  const int m0 = blockIdx.y * 128;
  const int n0 = blockIdx.x * 128;

  v8f acc[2][4] = {};

  for (int k0 = 0; k0 < EMB; k0 += 32) {
    __syncthreads();
    // ---- stage A tile 128x32 into LDS (f16) ----
    if constexpr (A_F16) {
      const _Float16* A = (const _Float16*)Ag;
#pragma unroll
      for (int i = 0; i < 2; ++i) {
        int id = tid + 256 * i;                 // 512 chunks of 8 halves
        int row = id >> 2, c8 = id & 3;
        h8 v = *(const h8*)(A + (size_t)(m0 + row) * EMB + k0 + c8 * 8);
        *(h8*)(sA + row * LDS_STRIDE + c8 * 8) = v;
      }
    } else {
      const float* A = (const float*)Ag;
#pragma unroll
      for (int i = 0; i < 4; ++i) {
        int id = tid + 256 * i;                 // 1024 chunks of 4 floats
        int row = id >> 3, c4 = id & 7;
        f4 v = *(const f4*)(A + (size_t)(m0 + row) * EMB + k0 + c4 * 4);
        h4 hv;
        hv[0] = (_Float16)v[0]; hv[1] = (_Float16)v[1];
        hv[2] = (_Float16)v[2]; hv[3] = (_Float16)v[3];
        *(h4*)(sA + row * LDS_STRIDE + c4 * 4) = hv;
      }
      // prefetch next k-tile of A (one cacheline probe per thread)
      if (k0 + 32 < EMB) {
        int row = tid >> 1, c = (tid & 1) * 16;
        __builtin_prefetch(A + (size_t)(m0 + row) * EMB + k0 + 32 + c, 0, 0);
      }
    }
    // ---- stage B tile: rows n0..n0+127 of W, cols k0..k0+31 (f32 -> f16) --
#pragma unroll
    for (int i = 0; i < 4; ++i) {
      int id = tid + 256 * i;
      int row = id >> 3, c4 = id & 7;
      f4 v = *(const f4*)(Wg + (size_t)(n0 + row) * EMB + k0 + c4 * 4);
      h4 hv;
      hv[0] = (_Float16)v[0]; hv[1] = (_Float16)v[1];
      hv[2] = (_Float16)v[2]; hv[3] = (_Float16)v[3];
      *(h4*)(sB + row * LDS_STRIDE + c4 * 4) = hv;
    }
    if (k0 + 32 < EMB) {                         // prefetch next W k-tile
      int row = tid >> 1, c = (tid & 1) * 16;
      __builtin_prefetch(Wg + (size_t)(n0 + row) * EMB + k0 + 32 + c, 0, 0);
    }
    __syncthreads();

    // ---- compute: 2x4 WMMA tiles per wave ----
    v16h bf[4];
#pragma unroll
    for (int nt = 0; nt < 4; ++nt)
      bf[nt] = load_frag_b(sB + (wn * 64 + nt * 16) * LDS_STRIDE, LDS_STRIDE, lane);
#pragma unroll
    for (int mt = 0; mt < 2; ++mt) {
      v16h af = load_frag_a(sA + (wm * 32 + mt * 16) * LDS_STRIDE, LDS_STRIDE, lane);
#pragma unroll
      for (int nt = 0; nt < 4; ++nt)
        acc[mt][nt] = wmma32(af, bf[nt], acc[mt][nt]);
    }
  }

  // ---- epilogue: C/D layout is (row = r + 8*(lane>=16), col = lane&15) ----
  const int hf = lane >> 4, ln = lane & 15;
#pragma unroll
  for (int mt = 0; mt < 2; ++mt)
#pragma unroll
    for (int nt = 0; nt < 4; ++nt)
#pragma unroll
      for (int r = 0; r < 8; ++r) {
        int row = m0 + wm * 32 + mt * 16 + r + 8 * hf;     // token s
        int col = n0 + wn * 64 + nt * 16 + ln;             // embed n
        float v = acc[mt][nt][r] * scale;
        if constexpr (OMODE == 0) {                        // [H][S][64]
          int h = col >> 6, d = col & 63;
          ((_Float16*)Og)[((size_t)h * SEQ + row) * HD + d] = (_Float16)v;
        } else if constexpr (OMODE == 1) {                 // [H][64][S]
          int h = col >> 6, d = col & 63;
          ((_Float16*)Og)[((size_t)h * HD + d) * SEQ + row] = (_Float16)v;
        } else {                                           // f32 [S][1024]
          ((float*)Og)[(size_t)row * EMB + col] = v;
        }
      }
}

// ---------------------------------------------------------------------------
// Flash attention, causal, per (head, 128-row slab).
// 256 thr = 8 waves; each wave owns 16 query rows and iterates only to its
// OWN diagonal (per-wave loop bound: no block-wide sync inside the loop,
// P staging is per-wave LDS).  Masking VALU runs only on diagonal strips.
// Softmax reductions use DPP16 ROW_XMASK (VALU) — LDS pipe stays free for
// the P-transpose staging and A-fragment loads.
// Q[H][S][64], K[H][S][64] (both pre-scaled by 1/8 => q.k/d), Vt[H][64][S].
// Online softmax == reference (softmax shift-invariant; mask == tril).
// ---------------------------------------------------------------------------
__global__ __launch_bounds__(256)
void flash_attn_kernel(const _Float16* __restrict__ Q,
                       const _Float16* __restrict__ Km,
                       const _Float16* __restrict__ Vt,
                       _Float16* __restrict__ Oa) {
  // per-wave P staging: 16 rows x 72 halves (padded) -> 2304 B, 16B aligned
  __shared__ __align__(16) _Float16 sP[8 * 16 * 72];

  const int tid = threadIdx.x, lane = tid & 31, wid = tid >> 5;
  const int h  = blockIdx.y;
  const int m0 = blockIdx.x * 128 + wid * 16;        // wave's first row
  const int hf = lane >> 4, ln = lane & 15;
  const float NEG_INF = -__builtin_inff();

  const _Float16* q  = Q  + (size_t)h * SEQ * HD;
  const _Float16* k  = Km + (size_t)h * SEQ * HD;
  const _Float16* vt = Vt + (size_t)h * HD * SEQ;
  _Float16* myP = sP + wid * 16 * 72;

  // Q fragments (K=64 -> two K=32 chunks), register-resident for whole loop
  v16h qf0 = load_frag_a(q + (size_t)m0 * HD, HD, lane);
  v16h qf1 = load_frag_a(q + (size_t)m0 * HD + 32, HD, lane);

  v8f o[4] = {};
  float mi[8], li[8];
#pragma unroll
  for (int r = 0; r < 8; ++r) { mi[r] = NEG_INF; li[r] = 0.f; }

  const int jEnd = m0 + 15;                          // wave's own diagonal
  for (int j0 = 0; j0 <= jEnd; j0 += 64) {
    // prefetch next strip of K while computing this one
    if (j0 + 64 <= jEnd)
      __builtin_prefetch(k + (size_t)(j0 + 64 + hf * 32) * HD + ln * 8, 0, 0);

    // ---- S = Q K^T over a 16x64 logits strip ----
    v8f st[4] = {};
#pragma unroll
    for (int nt = 0; nt < 4; ++nt) {
      const _Float16* kb = k + (size_t)(j0 + nt * 16) * HD;
      v16h b0 = load_frag_b(kb, HD, lane);
      v16h b1 = load_frag_b(kb + 32, HD, lane);
      st[nt] = wmma32(qf0, b0, st[nt]);
      st[nt] = wmma32(qf1, b1, st[nt]);
    }

    // ---- causal mask (only on diagonal-straddling strips) + row max ------
    float newm[8];
    if (j0 + 63 > m0) {                              // wave-uniform branch
#pragma unroll
      for (int r = 0; r < 8; ++r) {
        const int srow = m0 + r + 8 * hf;
        float mx = NEG_INF;
#pragma unroll
        for (int nt = 0; nt < 4; ++nt) {
          int col = j0 + nt * 16 + ln;
          float v = st[nt][r];
          if (col > srow) v = NEG_INF;               // causal
          st[nt][r] = v;
          mx = fmaxf(mx, v);
        }
        newm[r] = fmaxf(mi[r], rmax16(mx));
      }
    } else {                                         // fully-valid strip
#pragma unroll
      for (int r = 0; r < 8; ++r) {
        float mx = fmaxf(fmaxf(st[0][r], st[1][r]), fmaxf(st[2][r], st[3][r]));
        newm[r] = fmaxf(mi[r], rmax16(mx));
      }
    }

    // ---- p = exp(s - m), row sums, rescale running state ----
#pragma unroll
    for (int r = 0; r < 8; ++r) {
      float rs = 0.f;
#pragma unroll
      for (int nt = 0; nt < 4; ++nt) {
        float p = __expf(st[nt][r] - newm[r]);       // exp(-inf)=0 for masked
        st[nt][r] = p;
        rs += p;
      }
      rs = rsum16(rs);
      const float alpha = __expf(mi[r] - newm[r]);
      li[r] = li[r] * alpha + rs;
      mi[r] = newm[r];
#pragma unroll
      for (int nt = 0; nt < 4; ++nt) o[nt][r] *= alpha;   // rescale O
    }

    // ---- transpose P (C-layout -> A-layout) through per-wave LDS ----
#pragma unroll
    for (int nt = 0; nt < 4; ++nt)
#pragma unroll
      for (int r = 0; r < 8; ++r)
        myP[(r + 8 * hf) * 72 + nt * 16 + ln] = (_Float16)st[nt][r];
    // same-wave LDS RAW: drain DS counter before fragment reads
    asm volatile("s_wait_dscnt 0" ::: "memory");

    v16h p0 = load_frag_a(myP, 72, lane);            // K(j) = 0..31
    v16h p1 = load_frag_a(myP + 32, 72, lane);       // K(j) = 32..63

    // ---- O += P * V  (V pre-transposed: Vt rows = d, K = j contiguous) ----
#pragma unroll
    for (int nt = 0; nt < 4; ++nt) {
      const _Float16* vb = vt + (size_t)(nt * 16) * SEQ + j0;
      v16h b0 = load_frag_b(vb, SEQ, lane);
      v16h b1 = load_frag_b(vb + 32, SEQ, lane);
      o[nt] = wmma32(p0, b0, o[nt]);
      o[nt] = wmma32(p1, b1, o[nt]);
    }
  }

  // ---- normalize and emit attn (f16, [S][H*64]) for the output GEMM ----
#pragma unroll
  for (int nt = 0; nt < 4; ++nt)
#pragma unroll
    for (int r = 0; r < 8; ++r) {
      int srow = m0 + r + 8 * hf;
      float v = o[nt][r] / li[r];
      Oa[(size_t)srow * EMB + h * HD + nt * 16 + ln] = (_Float16)v;
    }
}

// ---------------------------------------------------------------------------
extern "C" void kernel_launch(void* const* d_in, const int* in_sizes, int n_in,
                              void* d_out, int out_size, void* d_ws, size_t ws_size,
                              hipStream_t stream) {
  const float* query = (const float*)d_in[0];
  const float* key_  = (const float*)d_in[1];
  const float* value = (const float*)d_in[2];
  // d_in[3] = tril mask: causality is computed analytically in-kernel
  const float* Wq = (const float*)d_in[4];
  const float* Wk = (const float*)d_in[5];
  const float* Wv = (const float*)d_in[6];
  const float* Wo = (const float*)d_in[7];

  // workspace: 4 x (2048*1024) f16 buffers = 16 MB
  _Float16* q16    = (_Float16*)d_ws;                  // [H][S][64], *1/8
  _Float16* k16    = q16 + (size_t)SEQ * EMB;          // [H][S][64], *1/8
  _Float16* vt16   = k16 + (size_t)SEQ * EMB;          // [H][64][S]
  _Float16* attn16 = vt16 + (size_t)SEQ * EMB;         // [S][1024]

  const float inv_sqrt_d = 0.125f;                     // 1/sqrt(64)
  dim3 blk(256);
  dim3 gG(EMB / 128, SEQ / 128);                       // (8,16) GEMM tiles
  dim3 gA(SEQ / 128, NH);                              // (16,16) attention

  gemm_xwT_kernel<false, 0><<<gG, blk, 0, stream>>>(query, Wq, q16,  inv_sqrt_d);
  gemm_xwT_kernel<false, 0><<<gG, blk, 0, stream>>>(key_,  Wk, k16,  inv_sqrt_d);
  gemm_xwT_kernel<false, 1><<<gG, blk, 0, stream>>>(value, Wv, vt16, 1.0f);
  flash_attn_kernel<<<gA, blk, 0, stream>>>(q16, k16, vt16, attn16);
  gemm_xwT_kernel<true, 2><<<gG, blk, 0, stream>>>(attn16, Wo, d_out, 1.0f);
}
